// NADE_52175262712368
// MI455X (gfx1250) — matrix operand order repaired and checked by reading, not compiled, not run on hardware
//
#include <hip/hip_runtime.h>

// NADE forward for MI455X (gfx1250, wave32, WMMA + async global->LDS).
// B=1024, D=784 sequential steps, H=512.
// 64 WGs x 16 batch rows; 4 waves/WG each owning a 128-wide hidden slice.
// Per step/wave: 4x v_wmma_f32_16x16x32_f16 (B = h_W row replicated over the
// 16 N columns), partial sums reduced across waves through LDS. The `a` state
// lives in VGPRs in WMMA A-fragment order. Staging of next step's h_W (f16)
// and in_W^T (f32) rows uses global_load_async_to_lds_b128 with double
// buffering -> exactly one s_barrier per step. Relu is done as packed f16 max
// after conversion (cvt output is canonical -> no v_max canonicalize pairs).

typedef __attribute__((ext_vector_type(16))) _Float16 v16h;
typedef __attribute__((ext_vector_type(8)))  float    v8f;

#define BATCH        1024
#define IN_DIM       784
#define HID          512
#define ROWS_PER_WG  16
#define WAVES_PER_WG 4
#define THREADS      (WAVES_PER_WG * 32)
#define K_PER_WAVE   (HID / WAVES_PER_WG)   // 128
#define CHUNKS       (K_PER_WAVE / 32)      // 4 WMMAs (K=32) per wave per step

// ---------------- pre-kernel: transpose in_W and convert h_W to f16 --------
__global__ void nade_prep(const float* __restrict__ inW,   // [H, D]
                          const float* __restrict__ hW,    // [D, H]
                          float* __restrict__ inWT,        // [D, H] f32
                          _Float16* __restrict__ hWh) {    // [D, H] f16
    int idx = blockIdx.x * blockDim.x + threadIdx.x;
    if (idx < IN_DIM * HID) {
        int i = idx / HID;
        int h = idx - i * HID;
        inWT[idx] = inW[(size_t)h * IN_DIM + i];   // coalesced writes
        hWh[idx]  = (_Float16)hW[idx];
    }
}

// ---------------- async copy helpers (CDNA5 ASYNCcnt path) -----------------
__device__ __forceinline__ void async_b128(unsigned lds_off, const void* gaddr) {
    // per-lane: LDS[lds_off .. +15] = MEM[gaddr .. +15], tracked by ASYNCcnt
    asm volatile("global_load_async_to_lds_b128 %0, %1, off"
                 :: "v"(lds_off), "v"(gaddr) : "memory");
}
__device__ __forceinline__ void wait_async0() {
    asm volatile("s_wait_asynccnt 0" ::: "memory");
}
__device__ __forceinline__ unsigned lds_addr(const void* p) {
    // generic pointer to __shared__: low 32 bits are the LDS offset (ISA 10.2)
    return (unsigned)(uintptr_t)p;
}

// relu(a) in f16 fragment form: cvt (canonical) then packed max with zero
__device__ __forceinline__ v16h relu_frag_f16(const float* a16) {
    v16h af;
#pragma unroll
    for (int e = 0; e < 16; ++e) af[e] = (_Float16)a16[e];
    v16h z = {};
    return __builtin_elementwise_max(af, z);   // v_pk_max_num_f16 x8
}

// ---------------- fast kernel: async-staged, f16 B, 1 barrier/step ---------
__global__ __launch_bounds__(THREADS)
void nade_fwd_async(const float* __restrict__ x,      // [B, D]
                    const float* __restrict__ uT,     // in_W^T [D, H] f32
                    const _Float16* __restrict__ wh,  // h_W [D, H] f16
                    const float* __restrict__ inb,    // [H]
                    const float* __restrict__ hb,     // [D]
                    float* __restrict__ out) {        // [B, D]
    __shared__ __align__(16) float    lds_u[2][HID];                  // 4 KB
    __shared__ __align__(32) _Float16 lds_wh[2][HID];                 // 2 KB
    __shared__ float lds_s[2][WAVES_PER_WG][ROWS_PER_WG];             // 0.5 KB

    const int tid     = threadIdx.x;
    const int wave    = tid >> 5;
    const int lane    = tid & 31;
    const int hiHalf  = (lane >= 16) ? 1 : 0;
    const int M       = lane & 15;
    const int rowBase = blockIdx.x * ROWS_PER_WG;
    const int row     = rowBase + M;

    // a-state in f32, in WMMA A-fragment element order:
    //   element e of lane L -> K = e + (e>=8?8:0) + (L>=16?8:0), M = L&15
    float aReg[CHUNKS][16];
#pragma unroll
    for (int c = 0; c < CHUNKS; ++c) {
#pragma unroll
        for (int e = 0; e < 16; ++e) {
            int kA = e + ((e >= 8) ? 8 : 0) + hiHalf * 8;
            aReg[c][e] = inb[wave * K_PER_WAVE + c * 32 + kA];
        }
    }

    // ---- stage step 0 into buffer 0 ----
    async_b128(lds_addr(&lds_u[0][tid * 4]), uT + tid * 4);          // 2 KB
    if (tid >= 64) {                                                  // 1 KB
        int t2 = tid - 64;
        async_b128(lds_addr(&lds_wh[0][t2 * 8]), wh + t2 * 8);
    }
    wait_async0();
    __syncthreads();

    for (int i = 0; i < IN_DIM; ++i) {
        const int cur = i & 1;
        const int nxt = cur ^ 1;

        // ---- kick off async staging of step i+1 (flies under the WMMAs) ----
        if (i + 1 < IN_DIM) {
            async_b128(lds_addr(&lds_u[nxt][tid * 4]),
                       uT + (size_t)(i + 1) * HID + tid * 4);
            if (tid >= 64) {
                int t2 = tid - 64;
                async_b128(lds_addr(&lds_wh[nxt][t2 * 8]),
                           wh + (size_t)(i + 1) * HID + t2 * 8);
            }
        }

        const float x_mi = x[(size_t)row * IN_DIM + i];

        // ---- s = relu(a_tile) @ w_i via 4x WMMA ----
        v8f cAcc = {};
#pragma unroll
        for (int c = 0; c < CHUNKS; ++c) {
            v16h af = relu_frag_f16(aReg[c]);
            // B fragment: K = e + (e>=8?8:0) + 8*laneHalf  ->  two 8 x f16 runs in LDS
            v16h bf;
#pragma unroll
            for (int e = 0; e < 16; ++e)
                bf[e] = lds_wh[cur][wave * K_PER_WAVE + c * 32 +
                                    e + ((e >= 8) ? 8 : 0) + hiHalf * 8];
            cAcc = __builtin_amdgcn_wmma_f32_16x16x32_f16(
                false, af, false, bf, (short)0, cAcc, false, false);
        }

        // C layout: VGPR r, lane L -> M = r + (L>=16?8:0); all N cols equal
        if (M == 0) {
#pragma unroll
            for (int r = 0; r < 8; ++r)
                lds_s[cur][wave][hiHalf * 8 + r] = cAcc[r];
        }

        // ---- rank-1 update: a += x_i * u_i ----
#pragma unroll
        for (int c = 0; c < CHUNKS; ++c) {
#pragma unroll
            for (int e = 0; e < 16; ++e) {
                int kA = e + ((e >= 8) ? 8 : 0) + hiHalf * 8;
                aReg[c][e] = fmaf(
                    x_mi, lds_u[cur][wave * K_PER_WAVE + c * 32 + kA],
                    aReg[c][e]);
            }
        }

        // ---- software-pipelined output: reduce column i-1 (barrier-safe) ----
        if (i > 0 && tid < ROWS_PER_WG) {
            float z = hb[i - 1];
#pragma unroll
            for (int w = 0; w < WAVES_PER_WG; ++w) z += lds_s[nxt][w][tid];
            out[(size_t)(rowBase + tid) * IN_DIM + (i - 1)] =
                1.0f / (1.0f + __expf(-z));
        }

        wait_async0();       // this wave's staged rows landed in LDS
        __syncthreads();     // single barrier/step: covers s, u, wh buffers
    }

    // ---- epilogue: last column ----
    if (tid < ROWS_PER_WG) {
        float z = hb[IN_DIM - 1];
#pragma unroll
        for (int w = 0; w < WAVES_PER_WG; ++w)
            z += lds_s[(IN_DIM - 1) & 1][w][tid];
        out[(size_t)(rowBase + tid) * IN_DIM + (IN_DIM - 1)] =
            1.0f / (1.0f + __expf(-z));
    }
}

// ---------------- fallback kernel (no workspace needed) --------------------
__global__ __launch_bounds__(THREADS)
void nade_fwd_sync(const float* __restrict__ x,
                   const float* __restrict__ uBase,  // in_W [H, D] (strided)
                   const float* __restrict__ inb,
                   const float* __restrict__ hW,     // [D, H]
                   const float* __restrict__ hb,
                   float* __restrict__ out) {
    __shared__ float lds_w[HID];
    __shared__ float lds_u[HID];
    __shared__ float lds_s[WAVES_PER_WG][ROWS_PER_WG];

    const int tid     = threadIdx.x;
    const int wave    = tid >> 5;
    const int lane    = tid & 31;
    const int hiHalf  = (lane >= 16) ? 1 : 0;
    const int M       = lane & 15;
    const int rowBase = blockIdx.x * ROWS_PER_WG;
    const int row     = rowBase + M;

    float aReg[CHUNKS][16];
#pragma unroll
    for (int c = 0; c < CHUNKS; ++c)
#pragma unroll
        for (int e = 0; e < 16; ++e) {
            int kA = e + ((e >= 8) ? 8 : 0) + hiHalf * 8;
            aReg[c][e] = inb[wave * K_PER_WAVE + c * 32 + kA];
        }

    for (int i = 0; i < IN_DIM; ++i) {
        const float* wsrc = hW + (size_t)i * HID;
#pragma unroll
        for (int j = 0; j < HID / THREADS; ++j) {
            int idx = tid + j * THREADS;
            lds_w[idx] = wsrc[idx];
            lds_u[idx] = uBase[(size_t)idx * IN_DIM + i];
        }
        __syncthreads();

        const float x_mi = x[(size_t)row * IN_DIM + i];

        v8f cAcc = {};
#pragma unroll
        for (int c = 0; c < CHUNKS; ++c) {
            v16h af = relu_frag_f16(aReg[c]);
            v16h bf;
#pragma unroll
            for (int e = 0; e < 16; ++e) {
                int kB = e + ((e >= 8) ? 8 : 0) + hiHalf * 8;
                bf[e] = (_Float16)lds_w[wave * K_PER_WAVE + c * 32 + kB];
            }
            cAcc = __builtin_amdgcn_wmma_f32_16x16x32_f16(
                false, af, false, bf, (short)0, cAcc, false, false);
        }

        if (M == 0) {
#pragma unroll
            for (int r = 0; r < 8; ++r)
                lds_s[wave][hiHalf * 8 + r] = cAcc[r];
        }

#pragma unroll
        for (int c = 0; c < CHUNKS; ++c)
#pragma unroll
            for (int e = 0; e < 16; ++e) {
                int kA = e + ((e >= 8) ? 8 : 0) + hiHalf * 8;
                aReg[c][e] = fmaf(
                    x_mi, lds_u[wave * K_PER_WAVE + c * 32 + kA], aReg[c][e]);
            }
        __syncthreads();

        if (tid < ROWS_PER_WG) {
            float z = hb[i];
#pragma unroll
            for (int w = 0; w < WAVES_PER_WG; ++w) z += lds_s[w][tid];
            out[(size_t)(rowBase + tid) * IN_DIM + i] =
                1.0f / (1.0f + __expf(-z));
        }
        __syncthreads();
    }
}

extern "C" void kernel_launch(void* const* d_in, const int* in_sizes, int n_in,
                              void* d_out, int out_size, void* d_ws, size_t ws_size,
                              hipStream_t stream) {
    const float* x   = (const float*)d_in[0];   // [1024, 784]
    const float* inW = (const float*)d_in[1];   // [512, 784]
    const float* inb = (const float*)d_in[2];   // [512]
    const float* hW  = (const float*)d_in[3];   // [784, 512]
    const float* hb  = (const float*)d_in[4];   // [784]
    float* out = (float*)d_out;                 // [1024, 784]

    const size_t n       = (size_t)IN_DIM * HID;
    const size_t tBytes  = n * sizeof(float);            // in_W^T
    const size_t hBytes  = n * sizeof(_Float16);         // h_W f16
    const size_t wsNeed  = tBytes + hBytes;              // ~2.3 MB

    if (d_ws != nullptr && ws_size >= wsNeed) {
        float*    inWT = (float*)d_ws;
        _Float16* hWh  = (_Float16*)((char*)d_ws + tBytes);
        nade_prep<<<((int)n + 255) / 256, 256, 0, stream>>>(inW, hW, inWT, hWh);
        nade_fwd_async<<<BATCH / ROWS_PER_WG, THREADS, 0, stream>>>(
            x, inWT, hWh, inb, hb, out);
    } else {
        nade_fwd_sync<<<BATCH / ROWS_PER_WG, THREADS, 0, stream>>>(
            x, inW, inb, hW, hb, out);
    }
}
